// CrossMambaBlock_42288247996525
// MI455X (gfx1250) — matrix operand order repaired and checked by reading, not compiled, not run on hardware
//
#include <hip/hip_runtime.h>
#include <hip/hip_bf16.h>

#define B_N   4
#define L_S   2304
#define C_D   768
#define D_IN  768
#define D_ST  16
#define DT_R  48
#define M_ROWS (B_N * L_S)   // 9216, divisible by 128

typedef __bf16 bf16_t;
typedef __attribute__((ext_vector_type(16))) __bf16 v16bf;
typedef __attribute__((ext_vector_type(8)))  __bf16 v8bf;
typedef __attribute__((ext_vector_type(8)))  float  v8f;
typedef __attribute__((ext_vector_type(4)))  int    v4i;

// Async global->LDS staging path (gfx1250). Falls back to direct loads if the
// builtin is unavailable on this toolchain.
#if __has_builtin(__builtin_amdgcn_global_load_async_to_lds_b128)
#define HAVE_ASYNC 1
#define AS1 __attribute__((address_space(1)))
#define AS3 __attribute__((address_space(3)))
#if __has_builtin(__builtin_amdgcn_s_wait_asynccnt)
#define WAIT_ASYNC() __builtin_amdgcn_s_wait_asynccnt(0)
#else
#define WAIT_ASYNC() asm volatile("s_wait_asynccnt 0x0" ::: "memory")
#endif
// 16B async copy: integer-route the casts (always-legal inttoptr) — a generic
// LDS pointer's low 32 bits are the LDS byte offset (aperture rule), and a
// generic global VA is the AS1 address verbatim.
static __device__ __forceinline__ void async_copy16(const void* g, void* l) {
    __builtin_amdgcn_global_load_async_to_lds_b128(
        (AS1 v4i*)(unsigned long long)(size_t)g,
        (AS3 v4i*)(unsigned int)(size_t)l, 0, 0);
}
#else
#define HAVE_ASYNC 0
#endif

// ---------------------------------------------------------------------------
// Weight conversion helpers (fp32 -> bf16, once per launch; deterministic)
// ---------------------------------------------------------------------------
__global__ void f32_to_bf16_kernel(const float* __restrict__ in, bf16_t* __restrict__ out, int n) {
    int i = blockIdx.x * 256 + threadIdx.x;
    if (i < n) out[i] = (bf16_t)in[i];
}

// dt_w is (768, 48); pad K to 64 with zeros so the WMMA K-loop stays 32-aligned
__global__ void pad48to64_bf16_kernel(const float* __restrict__ in, bf16_t* __restrict__ out, int rows) {
    int i = blockIdx.x * 256 + threadIdx.x;
    if (i >= rows * 64) return;
    int r = i >> 6, c = i & 63;
    out[i] = (c < DT_R) ? (bf16_t)in[r * DT_R + c] : (bf16_t)0.f;
}

// ---------------------------------------------------------------------------
// LayerNorm fused with scan-order gather: out[b][t][:] = LN(x[b][ idx(t) ][:])
// ---------------------------------------------------------------------------
__global__ void ln_gather_kernel(const float* __restrict__ x, const int* __restrict__ scan_idx,
                                 const float* __restrict__ g, const float* __restrict__ beta,
                                 bf16_t* __restrict__ out, int flip) {
    __shared__ float s_sum[256];
    __shared__ float s_sq[256];
    int row = blockIdx.x;                 // 0 .. B*L-1 (dest row, scan order)
    int b = row / L_S, t = row - b * L_S;
    int src = flip ? scan_idx[L_S - 1 - t] : scan_idx[t];
    const float* xr = x + ((size_t)b * L_S + src) * C_D;
    float lsum = 0.f, lsq = 0.f;
    for (int c = threadIdx.x; c < C_D; c += 256) { float v = xr[c]; lsum += v; lsq += v * v; }
    s_sum[threadIdx.x] = lsum; s_sq[threadIdx.x] = lsq;
    __syncthreads();
    for (int off = 128; off > 0; off >>= 1) {
        if (threadIdx.x < off) { s_sum[threadIdx.x] += s_sum[threadIdx.x + off];
                                 s_sq[threadIdx.x]  += s_sq[threadIdx.x + off]; }
        __syncthreads();
    }
    float mu   = s_sum[0] * (1.f / C_D);
    float var  = s_sq[0] * (1.f / C_D) - mu * mu;
    float rstd = rsqrtf(var + 1e-6f);
    bf16_t* orow = out + (size_t)row * C_D;
    for (int c = threadIdx.x; c < C_D; c += 256)
        orow[c] = (bf16_t)((xr[c] - mu) * rstd * g[c] + beta[c]);
}

// ---------------------------------------------------------------------------
// bf16 WMMA GEMM:  Out(M,N) = A(M,K) * W(N,K)^T   (+ fused epilogue)
// EPI 0: f32 store  | 1: softplus(v+bias) f32 | 2: gelu(v+bias) -> bf16
// EPI 3: v + bias + residual -> f32
// Wave tile 64x64 (16 WMMAs/k-step); block 128 threads = 4 waves (2x2) ->
// 128x128 block tile. A/B panels (128x32 each) are staged to LDS with
// double-buffered GLOBAL_LOAD_ASYNC_TO_LDS_B128 (ASYNCcnt), giving 64 FLOP/B
// of global traffic and feeding WMMA fragments from the DS pipe.
// ---------------------------------------------------------------------------
template <int EPI>
__device__ __forceinline__ void epi_store(void* Out, size_t idx, float v, float bv,
                                          const float* __restrict__ residual) {
    if constexpr (EPI == 0) {
        ((float*)Out)[idx] = v;
    } else if constexpr (EPI == 1) {
        v += bv;
        ((float*)Out)[idx] = (v > 20.f) ? v : log1pf(__expf(v));      // softplus
    } else if constexpr (EPI == 2) {
        v += bv;
        float gel = 0.5f * v * (1.f + erff(v * 0.70710678118654752f)); // exact GELU
        ((bf16_t*)Out)[idx] = (bf16_t)gel;
    } else {
        ((float*)Out)[idx] = v + bv + residual[idx];
    }
}

template <int EPI>
__global__ __launch_bounds__(128) void gemm_wmma_kernel(
    const bf16_t* __restrict__ A, const bf16_t* __restrict__ W, void* __restrict__ Out,
    const float* __restrict__ bias, const float* __restrict__ residual,
    int M, int N, int K) {
    const int lane = threadIdx.x & 31;
    const int wave = threadIdx.x >> 5;
    const int wm   = wave & 1;                    // M position within block (0..1)
    const int wn   = wave >> 1;                   // N position within block (0..1)
    const int half = lane >> 4;                   // K-half selector per ISA layout
    const int l16  = lane & 15;
    const int m0   = blockIdx.x * 128 + wm * 64;  // wave covers rows m0..m0+63
    const int ntb  = blockIdx.y * 8 + wn * 4;     // wave covers 16-wide N tiles ntb..ntb+3
    const int nTiles = (N + 15) >> 4;

    bool valid[4];
    #pragma unroll
    for (int j = 0; j < 4; ++j) valid[j] = (ntb + j) < nTiles;   // wave-uniform

    v8f acc[4][4];
    #pragma unroll
    for (int mi = 0; mi < 4; ++mi)
        #pragma unroll
        for (int j = 0; j < 4; ++j)
            #pragma unroll
            for (int r = 0; r < 8; ++r) acc[mi][j][r] = 0.f;

#if HAVE_ASYNC
    // Padded LDS panels: row stride 40 elems (80 B) => 16-lane fragment reads
    // land on distinct bank groups (20*r mod 64 is a permutation of 16 values).
    constexpr int LDR = 40;
    __shared__ bf16_t smA[2][128 * LDR];
    __shared__ bf16_t smB[2][128 * LDR];
    const int tid = threadIdx.x;
    const int aRow = blockIdx.x * 128 + tid;      // M < 9216 always in range
    int bRowN = blockIdx.y * 128 + tid;
    bRowN = (bRowN < N) ? bRowN : (N - 1);        // clamp; duplicates discarded
    const bf16_t* aSrc = A + (size_t)aRow * K;
    const bf16_t* wSrc = W + (size_t)bRowN * K;

    auto stage = [&](int buf, int k0) {
        bf16_t* dA = &smA[buf][tid * LDR];
        bf16_t* dB = &smB[buf][tid * LDR];
        #pragma unroll
        for (int j = 0; j < 4; ++j) {
            async_copy16(aSrc + k0 + j * 8, dA + j * 8);
            async_copy16(wSrc + k0 + j * 8, dB + j * 8);
        }
    };

    stage(0, 0);
    WAIT_ASYNC();
    __syncthreads();

    const int KT = K / 32;
    #pragma unroll 1
    for (int kt = 0; kt < KT; ++kt) {
        const int cur = kt & 1;
        if (kt + 1 < KT) stage(1 - cur, (kt + 1) * 32);   // overlap with compute

        union { v16bf v; v8bf h[2]; } a[4], b[4];
        #pragma unroll
        for (int mi = 0; mi < 4; ++mi) {
            const bf16_t* ar = &smA[cur][(wm * 64 + mi * 16 + l16) * LDR];
            a[mi].h[0] = *(const v8bf*)(ar + half * 8);
            a[mi].h[1] = *(const v8bf*)(ar + 16 + half * 8);
        }
        #pragma unroll
        for (int j = 0; j < 4; ++j) {
            const bf16_t* br = &smB[cur][((wn * 4 + j) * 16 + l16) * LDR];
            b[j].h[0] = *(const v8bf*)(br + half * 8);
            b[j].h[1] = *(const v8bf*)(br + 16 + half * 8);
        }

        #pragma unroll
        for (int mi = 0; mi < 4; ++mi)
            #pragma unroll
            for (int j = 0; j < 4; ++j)
                acc[mi][j] = __builtin_amdgcn_wmma_f32_16x16x32_bf16(
                    false, a[mi].v, false, b[j].v, (short)0, acc[mi][j], false, false);

        WAIT_ASYNC();          // staged panel complete before next iteration
        __syncthreads();       // all waves done reading buf(cur) before restage
    }
#else
    // Fallback: direct-from-global fragments (proven path)
    const bf16_t* aBase[4];
    #pragma unroll
    for (int mi = 0; mi < 4; ++mi)
        aBase[mi] = A + (size_t)(m0 + mi * 16 + l16) * K + half * 8;
    const bf16_t* wBase[4];
    #pragma unroll
    for (int j = 0; j < 4; ++j) {
        int ntc  = valid[j] ? (ntb + j) : 0;
        wBase[j] = W + (size_t)(ntc * 16 + l16) * K + half * 8;
    }
    #pragma unroll 1
    for (int k0 = 0; k0 < K; k0 += 32) {
        int kpf = k0 + 64;
        kpf = (kpf < K) ? kpf : 0;
        __builtin_prefetch(aBase[0] + kpf, 0, 1);
        __builtin_prefetch(wBase[0] + kpf, 0, 1);
        union { v16bf v; v8bf h[2]; } a[4];
        union { v16bf v; v8bf h[2]; } b[4];
        #pragma unroll
        for (int mi = 0; mi < 4; ++mi) {
            a[mi].h[0] = *(const v8bf*)(aBase[mi] + k0);
            a[mi].h[1] = *(const v8bf*)(aBase[mi] + k0 + 16);
        }
        #pragma unroll
        for (int j = 0; j < 4; ++j) {
            b[j].h[0] = *(const v8bf*)(wBase[j] + k0);
            b[j].h[1] = *(const v8bf*)(wBase[j] + k0 + 16);
        }
        #pragma unroll
        for (int mi = 0; mi < 4; ++mi)
            #pragma unroll
            for (int j = 0; j < 4; ++j)
                acc[mi][j] = __builtin_amdgcn_wmma_f32_16x16x32_bf16(
                    false, a[mi].v, false, b[j].v, (short)0, acc[mi][j], false, false);
    }
#endif

    // C/D layout: lanes 0-15 rows +0..7, lanes 16-31 rows +8..15; col = tile*16 + (lane&15)
    #pragma unroll
    for (int mi = 0; mi < 4; ++mi) {
        const int rbase = m0 + mi * 16 + half * 8;
        #pragma unroll
        for (int j = 0; j < 4; ++j) {
            if (!valid[j]) continue;
            int col = (ntb + j) * 16 + l16;
            if (col >= N) continue;
            float bv = 0.f;
            if constexpr (EPI != 0) bv = bias[col];
            #pragma unroll
            for (int r = 0; r < 8; ++r) {
                size_t idx = (size_t)(rbase + r) * N + col;
                epi_store<EPI>(Out, idx, acc[mi][j][r], bv, residual);
            }
        }
    }
}

// ---------------------------------------------------------------------------
// Depthwise causal conv (width 4) + bias + SiLU on xi = xz[:, :768]
// ---------------------------------------------------------------------------
__global__ void conv_silu_kernel(const float* __restrict__ xz, const float* __restrict__ conv_w,
                                 const float* __restrict__ conv_b, float* __restrict__ xc,
                                 bf16_t* __restrict__ xc_bf) {
    int t = blockIdx.x, b = blockIdx.y;
    for (int d = threadIdx.x; d < D_IN; d += 256) {
        float acc = conv_b[d];
        #pragma unroll
        for (int k = 0; k < 4; ++k) {
            int tt = t - 3 + k;
            if (tt >= 0) acc += conv_w[d * 4 + k] * xz[((size_t)(b * L_S + tt)) * (2 * D_IN) + d];
        }
        float s = acc / (1.f + __expf(-acc));
        size_t o = ((size_t)(b * L_S + t)) * D_IN + d;
        xc[o] = s;
        xc_bf[o] = (bf16_t)s;
    }
}

// stage dt-rank columns of x_dbl (cols 0..47 of 80) into K=64 zero-padded bf16
__global__ void stage_dtr_kernel(const float* __restrict__ xdbl, bf16_t* __restrict__ dtr, int n) {
    int i = blockIdx.x * 256 + threadIdx.x;
    if (i >= n) return;
    int r = i >> 6, c = i & 63;
    dtr[i] = (c < DT_R) ? (bf16_t)xdbl[r * 80 + c] : (bf16_t)0.f;
}

// ---------------------------------------------------------------------------
// Selective scan: one lane per (b,d) channel, h[16]/A[16] in registers.
// Fuses + Dp*xc and the * silu(z) gate; emits bf16 for out_proj WMMA.
// ---------------------------------------------------------------------------
__global__ void scan_kernel(const float* __restrict__ dt, const float* __restrict__ xdbl,
                            const float* __restrict__ xc, const float* __restrict__ xz,
                            const float* __restrict__ A_log, const float* __restrict__ Dp,
                            bf16_t* __restrict__ ym) {
    int ch = blockIdx.x * 256 + threadIdx.x;        // 0 .. B*D_IN-1
    int b = ch / D_IN, d = ch - b * D_IN;
    float a[D_ST], h[D_ST];
    #pragma unroll
    for (int s = 0; s < D_ST; ++s) { a[s] = -__expf(A_log[d * D_ST + s]); h[s] = 0.f; }
    float dpv = Dp[d];
    for (int t = 0; t < L_S; ++t) {
        size_t row = (size_t)b * L_S + t;
        float dtv = dt[row * D_IN + d];
        float xv  = xc[row * D_IN + d];
        const float* bc = xdbl + row * 80 + DT_R;   // B at [0..15], C at [16..31]
        float dx = dtv * xv;
        float y = 0.f;
        #pragma unroll
        for (int s = 0; s < D_ST; ++s) {
            h[s] = __expf(dtv * a[s]) * h[s] + dx * bc[s];
            y += h[s] * bc[D_ST + s];
        }
        y += dpv * xv;
        float zv = xz[row * (2 * D_IN) + D_IN + d];
        y *= zv / (1.f + __expf(-zv));              // * silu(z)
        ym[row * D_IN + d] = (bf16_t)y;
    }
}

// inverse-permute branch output into concatenated fused buffer (bf16 half)
__global__ void scatter_bf16_kernel(const float* __restrict__ src, const int* __restrict__ scan_idx,
                                    bf16_t* __restrict__ fused, int flip, int halfOff, long n) {
    long i = (long)blockIdx.x * 256 + threadIdx.x;  // B*L*768 elements
    if (i >= n) return;
    int c = (int)(i % D_IN);
    long bt = i / D_IN;
    int t = (int)(bt % L_S), b = (int)(bt / L_S);
    int pos = flip ? scan_idx[L_S - 1 - t] : scan_idx[t];
    fused[(((size_t)b * L_S + pos) * (2 * C_D)) + halfOff + c] = (bf16_t)src[i];
}

// ---------------------------------------------------------------------------
// Host orchestration
// ---------------------------------------------------------------------------
extern "C" void kernel_launch(void* const* d_in, const int* in_sizes, int n_in,
                              void* d_out, int out_size, void* d_ws, size_t ws_size,
                              hipStream_t stream) {
    (void)in_sizes; (void)n_in; (void)out_size; (void)ws_size;
    const float* x        = (const float*)d_in[0];
    const int*   scan_idx = (const int*)  d_in[1];
    const float* norm_g   = (const float*)d_in[2];
    const float* norm_b   = (const float*)d_in[3];
    const float* fuse_w1  = (const float*)d_in[4];
    const float* fuse_b1  = (const float*)d_in[5];
    const float* fuse_w2  = (const float*)d_in[6];
    const float* fuse_b2  = (const float*)d_in[7];
    auto P = [&](int i) { return (const float*)d_in[i]; };
    // f_: 8 in_w, 9 conv_w, 10 conv_b, 11 xproj_w, 12 dt_w, 13 dt_b, 14 A_log, 15 Dp, 16 out_w
    // b_: 17..25 in the same order

    // workspace arena (bump allocator, 256B aligned)
    char* base = (char*)d_ws;
    size_t off = 0;
    auto alloc = [&](size_t bytes) -> void* {
        off = (off + 255) & ~(size_t)255;
        void* p = base + off; off += bytes; return p;
    };
    bf16_t* w_in_f  = (bf16_t*)alloc((size_t)2 * D_IN * C_D * 2);
    bf16_t* w_in_b  = (bf16_t*)alloc((size_t)2 * D_IN * C_D * 2);
    bf16_t* w_xp_f  = (bf16_t*)alloc((size_t)80 * D_IN * 2);
    bf16_t* w_xp_b  = (bf16_t*)alloc((size_t)80 * D_IN * 2);
    bf16_t* w_dt_f  = (bf16_t*)alloc((size_t)D_IN * 64 * 2);
    bf16_t* w_dt_b  = (bf16_t*)alloc((size_t)D_IN * 64 * 2);
    bf16_t* w_out_f = (bf16_t*)alloc((size_t)C_D * D_IN * 2);
    bf16_t* w_out_b = (bf16_t*)alloc((size_t)C_D * D_IN * 2);
    bf16_t* w_f1    = (bf16_t*)alloc((size_t)C_D * 2 * C_D * 2);
    bf16_t* w_f2    = (bf16_t*)alloc((size_t)C_D * C_D * 2);
    bf16_t* xg_bf   = (bf16_t*)alloc((size_t)M_ROWS * C_D * 2);
    float*  xz      = (float*) alloc((size_t)M_ROWS * 2 * D_IN * 4);
    float*  xc      = (float*) alloc((size_t)M_ROWS * D_IN * 4);
    bf16_t* xc_bf   = (bf16_t*)alloc((size_t)M_ROWS * D_IN * 2);
    float*  x_dbl   = (float*) alloc((size_t)M_ROWS * 80 * 4);
    bf16_t* dtr_bf  = (bf16_t*)alloc((size_t)M_ROWS * 64 * 2);
    float*  dt_buf  = (float*) alloc((size_t)M_ROWS * D_IN * 4);
    bf16_t* ym_bf   = (bf16_t*)alloc((size_t)M_ROWS * D_IN * 2);
    float*  g4out   = (float*) alloc((size_t)M_ROWS * D_IN * 4);
    bf16_t* fused_bf= (bf16_t*)alloc((size_t)M_ROWS * 2 * C_D * 2);
    bf16_t* h_bf    = (bf16_t*)alloc((size_t)M_ROWS * C_D * 2);

    auto cvt = [&](const float* src, bf16_t* dst, int n) {
        f32_to_bf16_kernel<<<(n + 255) / 256, 256, 0, stream>>>(src, dst, n);
    };
    cvt(P(8),  w_in_f,  2 * D_IN * C_D);
    cvt(P(17), w_in_b,  2 * D_IN * C_D);
    cvt(P(11), w_xp_f,  80 * D_IN);
    cvt(P(20), w_xp_b,  80 * D_IN);
    pad48to64_bf16_kernel<<<(D_IN * 64 + 255) / 256, 256, 0, stream>>>(P(12), w_dt_f, D_IN);
    pad48to64_bf16_kernel<<<(D_IN * 64 + 255) / 256, 256, 0, stream>>>(P(21), w_dt_b, D_IN);
    cvt(P(16), w_out_f, C_D * D_IN);
    cvt(P(25), w_out_b, C_D * D_IN);
    cvt(fuse_w1, w_f1, C_D * 2 * C_D);
    cvt(fuse_w2, w_f2, C_D * C_D);

    const dim3 blk(256);
    const dim3 gblk(128);          // GEMM blocks: 4 waves (2x2), 128x128 tile
    const int GM = M_ROWS / 128;   // 72 block rows
    auto gy = [](int N) { return (((N + 15) >> 4) + 7) / 8; };  // 128 N per block
    const long nScatter = (long)M_ROWS * D_IN;

    auto run_branch = [&](int flip, const bf16_t* w_in, const float* conv_w, const float* conv_b,
                          const bf16_t* w_xp, const bf16_t* w_dt, const float* dt_b,
                          const float* A_log, const float* Dp, const bf16_t* w_out, int halfOff) {
        ln_gather_kernel<<<M_ROWS, blk, 0, stream>>>(x, scan_idx, norm_g, norm_b, xg_bf, flip);
        // in_proj: (9216 x 1536) = xg * in_w^T
        gemm_wmma_kernel<0><<<dim3(GM, gy(2 * D_IN)), gblk, 0, stream>>>(
            xg_bf, w_in, xz, nullptr, nullptr, M_ROWS, 2 * D_IN, C_D);
        conv_silu_kernel<<<dim3(L_S, B_N), blk, 0, stream>>>(xz, conv_w, conv_b, xc, xc_bf);
        // x_proj: (9216 x 80)
        gemm_wmma_kernel<0><<<dim3(GM, gy(DT_R + 2 * D_ST)), gblk, 0, stream>>>(
            xc_bf, w_xp, x_dbl, nullptr, nullptr, M_ROWS, DT_R + 2 * D_ST, C_D);
        stage_dtr_kernel<<<(M_ROWS * 64 + 255) / 256, blk, 0, stream>>>(x_dbl, dtr_bf, M_ROWS * 64);
        // dt_proj + softplus(+dt_b): (9216 x 768), K padded to 64
        gemm_wmma_kernel<1><<<dim3(GM, gy(D_IN)), gblk, 0, stream>>>(
            dtr_bf, w_dt, dt_buf, dt_b, nullptr, M_ROWS, D_IN, 64);
        scan_kernel<<<(B_N * D_IN) / 256, blk, 0, stream>>>(dt_buf, x_dbl, xc, xz, A_log, Dp, ym_bf);
        // out_proj: (9216 x 768)
        gemm_wmma_kernel<0><<<dim3(GM, gy(C_D)), gblk, 0, stream>>>(
            ym_bf, w_out, g4out, nullptr, nullptr, M_ROWS, C_D, D_IN);
        scatter_bf16_kernel<<<(int)((nScatter + 255) / 256), blk, 0, stream>>>(
            g4out, scan_idx, fused_bf, flip, halfOff, nScatter);
    };

    run_branch(0, w_in_f, P(9),  P(10), w_xp_f, w_dt_f, P(13), P(14), P(15), w_out_f, 0);
    run_branch(1, w_in_b, P(18), P(19), w_xp_b, w_dt_b, P(22), P(23), P(24), w_out_b, C_D);

    // fuse1 + GELU -> bf16: (9216 x 768) = fused(9216x1536) * fuse_w1^T
    gemm_wmma_kernel<2><<<dim3(GM, gy(C_D)), gblk, 0, stream>>>(
        fused_bf, w_f1, h_bf, fuse_b1, nullptr, M_ROWS, C_D, 2 * C_D);
    // fuse2 + bias + residual -> f32 output
    gemm_wmma_kernel<3><<<dim3(GM, gy(C_D)), gblk, 0, stream>>>(
        h_bf, w_f2, d_out, fuse_b2, x, M_ROWS, C_D, C_D);
}
